// GraphAttentionNetwork_70531952935242
// MI455X (gfx1250) — matrix operand order repaired and checked by reading, not compiled, not run on hardware
//
#include <hip/hip_runtime.h>
#include <hip/hip_bf16.h>

// ---------------------------------------------------------------------------
// GAT layer for MI455X (gfx1250, wave32, WMMA).
//   N=8192, IN_F=256, OUT_F=128.
//   Roofline: kernel is HBM-bound on the 256MB adj read (~11us @ 23.3TB/s).
//   -> single fused flash-softmax pass (never materialize the 8192^2 att),
//      P@h via v_wmma_f32_16x16x32_bf16 (fp32 accum), h (2MB bf16) L2-resident.
//   All global traffic is b128-vectorized; fragments load as ds_load_b128.
// ---------------------------------------------------------------------------

#define NN      8192
#define IN_F    256
#define OF      128
#define TJ      64          // column tile = 2 x WMMA-K(32)
#define HPAD    80          // padded K-extent of transposed h tile (160B rows)
#define NEG_BIG (-9.0e15f)

typedef __attribute__((ext_vector_type(16))) __bf16 v16bf;
typedef __attribute__((ext_vector_type(8)))  float  v8f;

__device__ __forceinline__ unsigned short f32_to_bf16u(float f) {
    union { float f; unsigned u; } x; x.f = f;
    unsigned u = x.u;
    u += 0x7fffu + ((u >> 16) & 1u);          // round to nearest even
    return (unsigned short)(u >> 16);
}

// Build a v16bf WMMA fragment from two contiguous 16-byte LDS runs.
__device__ __forceinline__ v16bf load_frag(const unsigned short* p0,
                                           const unsigned short* p1) {
    union { uint4 u[2]; v16bf v; } f;
    f.u[0] = *(const uint4*)p0;
    f.u[1] = *(const uint4*)p1;
    return f.v;
}

// ---------------------------------------------------------------------------
// Kernel 1: h = x@W + b (row per block) -> bf16 h, s1 = h.a1, s2 = h.a2 + ba.
// ~0.54 GFLOP total, negligible next to the adj stream; plain VALU.
// ---------------------------------------------------------------------------
__global__ __launch_bounds__(128)
void gat_fc_kernel(const float* __restrict__ x,  const float* __restrict__ W,
                   const float* __restrict__ b,  const float* __restrict__ a1,
                   const float* __restrict__ a2, const float* __restrict__ ba,
                   unsigned short* __restrict__ hb,
                   float* __restrict__ s1, float* __restrict__ s2)
{
    __shared__ float r1[128];
    __shared__ float r2[128];
    const int i = blockIdx.x;
    const int f = threadIdx.x;

    const float* xr = x + (size_t)i * IN_F;
    float acc = b[f];
#pragma unroll 4
    for (int k = 0; k < IN_F; ++k)
        acc += xr[k] * W[k * OF + f];

    hb[(size_t)i * OF + f] = f32_to_bf16u(acc);
    r1[f] = acc * a1[f];
    r2[f] = acc * a2[f];
    __syncthreads();
#pragma unroll
    for (int s = 64; s > 0; s >>= 1) {
        if (f < s) { r1[f] += r1[f + s]; r2[f] += r2[f + s]; }
        __syncthreads();
    }
    if (f == 0) { s1[i] = r1[0]; s2[i] = r2[0] + ba[0]; }
}

// ---------------------------------------------------------------------------
// Kernel 2: fused masked-softmax attention, flash style.
// Block = 256 threads (8 wave32) handles 16 output rows.
// Wave w owns output columns [16w, 16w+16): one v8f WMMA accumulator/wave.
// Per j-tile (64 cols): one int4 adj load + one float4 s2 load per thread,
// shfl-xor online softmax, packed b64 P stores, b128-staged transposed h,
// then two 16x16x32 bf16 WMMAs.
// ---------------------------------------------------------------------------
__global__ __launch_bounds__(256)
void gat_attn_kernel(const int* __restrict__ adj,
                     const unsigned short* __restrict__ hb,
                     const float* __restrict__ s1,
                     const float* __restrict__ s2,
                     float* __restrict__ out)
{
    __shared__ __align__(16) unsigned short pT[16][TJ];    // bf16 P tile (A)
    __shared__ __align__(16) unsigned short hTt[OF][HPAD]; // bf16 h tile, transposed (B)
    __shared__ float mRow[16], lRow[16], sRow[16], s1Row[16];

    const int tid  = threadIdx.x;
    const int lane = tid & 31;
    const int wave = tid >> 5;
    const int hi   = lane >> 4;     // 16-lane half within the wave
    const int lo   = lane & 15;
    const int n0   = wave * 16;     // this wave's output-column base
    const int i0   = blockIdx.x * 16;

    const int r = tid >> 4;         // row 0..15 (16 threads per row)
    const int s = tid & 15;         // position within the row group
    const int c0 = s * 4;           // this thread's 4 consecutive columns

    if (tid < 16) {
        mRow[tid]  = NEG_BIG;
        lRow[tid]  = 0.0f;
        s1Row[tid] = s1[i0 + tid];
    }
    v8f acc = {};
    __syncthreads();

    const int* arow = adj + (size_t)(i0 + r) * NN;

    for (int j0 = 0; j0 < NN; j0 += TJ) {
        // ---- stage h tile transposed: 4 x global_load_b128 per thread -----
        const uint4* hbv = (const uint4*)(hb + (size_t)j0 * OF);
#pragma unroll
        for (int q = 0; q < 4; ++q) {
            int idx = tid + q * 256;           // 0..1023 = 64 rows x 16 chunks
            int kk  = idx >> 4;                // 0..63   (k within tile)
            int c8  = (idx & 15) * 8;          // feature base (coalesced)
            union { uint4 v; unsigned short u[8]; } d;
            d.v = hbv[(size_t)kk * (OF / 8) + (idx & 15)];
#pragma unroll
            for (int m = 0; m < 8; ++m) hTt[c8 + m][kk] = d.u[m];
        }

        // ---- scores + online softmax (vector loads, shfl reductions) ------
        __builtin_prefetch(arow + j0 + TJ, 0, 1);   // next adj tile (stream)
        const int4   a4 = *(const int4*)  (arow + j0 + c0);
        const float4 t4 = *(const float4*)(s2   + j0 + c0);
        const float mOld = mRow[r];
        const float sv   = s1Row[r];
        float e[4];
        {
            float v0 = sv + t4.x, v1 = sv + t4.y, v2 = sv + t4.z, v3 = sv + t4.w;
            e[0] = (a4.x > 0) ? ((v0 > 0.f) ? v0 : 0.2f * v0) : NEG_BIG;
            e[1] = (a4.y > 0) ? ((v1 > 0.f) ? v1 : 0.2f * v1) : NEG_BIG;
            e[2] = (a4.z > 0) ? ((v2 > 0.f) ? v2 : 0.2f * v2) : NEG_BIG;
            e[3] = (a4.w > 0) ? ((v3 > 0.f) ? v3 : 0.2f * v3) : NEG_BIG;
        }
        float mx = fmaxf(fmaxf(e[0], e[1]), fmaxf(e[2], e[3]));
        mx = fmaxf(mx, mOld);
#pragma unroll
        for (int off = 8; off >= 1; off >>= 1)
            mx = fmaxf(mx, __shfl_xor(mx, off, 16));

        float p0 = __expf(e[0] - mx), p1 = __expf(e[1] - mx);
        float p2 = __expf(e[2] - mx), p3 = __expf(e[3] - mx);
        uint2 pk;
        pk.x = (unsigned)f32_to_bf16u(p0) | ((unsigned)f32_to_bf16u(p1) << 16);
        pk.y = (unsigned)f32_to_bf16u(p2) | ((unsigned)f32_to_bf16u(p3) << 16);
        *(uint2*)&pT[r][c0] = pk;                   // one ds_store_b64

        float sum = (p0 + p1) + (p2 + p3);
#pragma unroll
        for (int off = 8; off >= 1; off >>= 1)
            sum += __shfl_xor(sum, off, 16);
        if (s == 0) {                    // rows 2w,2w+1 owned by wave w: no race
            float sc = __expf(mOld - mx);
            sRow[r] = sc;
            lRow[r] = lRow[r] * sc + sum;
            mRow[r] = mx;
        }
        __syncthreads();

        // ---- rescale accumulator, two WMMAs over K=64 ---------------------
#pragma unroll
        for (int g = 0; g < 8; ++g)
            acc[g] *= sRow[g + 8 * hi];            // C/D: VGPR g -> row g+8*hi

        // A 16x32 bf16: lane=row; elems = K 16hi..16hi+15 (contiguous in pT)
        // B 32x16 bf16: lane=col; elems = K 16hi..16hi+15 (contiguous in hTt)
        v16bf aF0 = load_frag(&pT[lo][16 * hi],        &pT[lo][16 * hi + 8]);
        v16bf bF0 = load_frag(&hTt[n0 + lo][16 * hi], &hTt[n0 + lo][16 * hi + 8]);
        acc = __builtin_amdgcn_wmma_f32_16x16x32_bf16(
                  false, aF0, false, bF0, (short)0, acc, false, false);

        v16bf aF1 = load_frag(&pT[lo][32 + 16 * hi],        &pT[lo][32 + 16 * hi + 8]);
        v16bf bF1 = load_frag(&hTt[n0 + lo][32 + 16 * hi], &hTt[n0 + lo][32 + 16 * hi + 8]);
        acc = __builtin_amdgcn_wmma_f32_16x16x32_bf16(
                  false, aF1, false, bF1, (short)0, acc, false, false);

        __syncthreads();   // protect LDS tiles before next iteration's writes
    }

    // ---- epilogue: divide by softmax denominator, write fp32 output -------
#pragma unroll
    for (int g = 0; g < 8; ++g) {
        int m = g + 8 * hi;
        out[(size_t)(i0 + m) * OF + (n0 + lo)] = acc[g] / lRow[m];
    }
}

// ---------------------------------------------------------------------------
extern "C" void kernel_launch(void* const* d_in, const int* in_sizes, int n_in,
                              void* d_out, int out_size, void* d_ws, size_t ws_size,
                              hipStream_t stream)
{
    (void)in_sizes; (void)n_in; (void)out_size; (void)ws_size;

    const float* x   = (const float*)d_in[0];
    const int*   adj = (const int*)  d_in[1];
    const float* W   = (const float*)d_in[2];
    const float* b   = (const float*)d_in[3];
    const float* a1  = (const float*)d_in[4];
    const float* a2  = (const float*)d_in[5];
    const float* ba  = (const float*)d_in[6];
    float* out = (float*)d_out;

    // workspace: bf16 h (2 MB) + s1 (32 KB) + s2 (32 KB)
    unsigned short* hb = (unsigned short*)d_ws;
    float* s1 = (float*)((char*)d_ws + (size_t)NN * OF * sizeof(unsigned short));
    float* s2 = s1 + NN;

    gat_fc_kernel<<<NN, 128, 0, stream>>>(x, W, b, a1, a2, ba, hb, s1, s2);
    gat_attn_kernel<<<NN / 16, 256, 0, stream>>>(adj, hb, s1, s2, out);
}
